// SocialLSTM_64476049048304
// MI455X (gfx1250) — matrix operand order repaired and checked by reading, not compiled, not run on hardware
//
#include <hip/hip_runtime.h>

#define SEQ 20
#define NN  512
#define RNN 128
#define EMB 64
#define G2  16
#define OUTD 5

typedef __attribute__((ext_vector_type(16))) __bf16 v16bf;
typedef __attribute__((ext_vector_type(2)))  __bf16 v2bf;
typedef __attribute__((ext_vector_type(8)))  float  v8f;

union V16 { v16bf v; unsigned short u[16]; unsigned int w[8]; };
struct alignas(16) US8  { unsigned short s[8]; };
struct alignas(32) US16 { unsigned short s[16]; };
union U8pack { US8 us; unsigned int w[4]; };

__device__ __forceinline__ unsigned short f2bf(float f) {
  union { float f; unsigned int i; } c; c.f = f;
  unsigned int x = c.i + 0x7FFFu + ((c.i >> 16) & 1u);  // RNE
  return (unsigned short)(x >> 16);
}

// Convert+pack two floats into one dword of two bf16 (RNE).
__device__ __forceinline__ unsigned int pack2bf(float lo, float hi) {
#if __has_builtin(__builtin_amdgcn_cvt_pk_bf16_f32)
  v2bf p = __builtin_amdgcn_cvt_pk_bf16_f32(lo, hi);
  return __builtin_bit_cast(unsigned int, p);
#else
  union { float f; unsigned int i; } a, b; a.f = lo; b.f = hi;
  unsigned int xl = a.i + 0x7FFFu + ((a.i >> 16) & 1u);
  unsigned int xh = b.i + 0x7FFFu + ((b.i >> 16) & 1u);
  return (xl >> 16) | (xh & 0xFFFF0000u);
#endif
}

__device__ __forceinline__ v8f zero8() {
  v8f z = {0.f, 0.f, 0.f, 0.f, 0.f, 0.f, 0.f, 0.f};
  return z;
}

__device__ __forceinline__ v8f wmma_bf16(const V16& a, const V16& b, v8f c) {
  return __builtin_amdgcn_wmma_f32_16x16x32_bf16(
      false, a.v, false, b.v, (short)0, c, false, false);
}

// A-tile from contiguous bf16 row. ISA 16-bit A layout:
// lane half=0 holds K={0..7,16..23}, half=1 holds K={8..15,24..31}.
__device__ __forceinline__ V16 loadA_row(const unsigned short* rowbase, int k0, int half) {
  V16 a;
  *(US8*)&a.u[0] = *(const US8*)&rowbase[k0 + half * 8];
  *(US8*)&a.u[8] = *(const US8*)&rowbase[k0 + 16 + half * 8];
  return a;
}

// B-tile (32x16): lane holds output-column, K = {half*8..+7, 16+half*8..+7} (matches A).
__device__ __forceinline__ V16 loadB_row(const unsigned short* base, size_t rowoff,
                                         int k0, int half) {
  V16 b;
  *(US8*)&b.u[0] = *(const US8*)&base[rowoff + k0 + half * 8];
  *(US8*)&b.u[8] = *(const US8*)&base[rowoff + k0 + 16 + half * 8];
  return b;
}

// Strided A gather from grids: A[g][m] = G[m][g], f32 -> packed bf16.
__device__ __forceinline__ V16 gatherA_grid(const float* __restrict__ G, int k0,
                                            int half, int row) {
  V16 a;
#pragma unroll
  for (int i = 0; i < 8; ++i) {
    int j = 2 * i;
    int kj = (j & 7) + half * 8 + ((j >> 3) << 4);
    float lo = G[(size_t)(k0 + kj) * G2 + row];
    float hi = G[(size_t)(k0 + kj + 1) * G2 + row];
    a.w[i] = pack2bf(lo, hi);
  }
  return a;
}

__global__ void init_kernel(const float* __restrict__ hidden, const float* __restrict__ cell,
                            const float* __restrict__ W_te, const float* __restrict__ W_ih,
                            const float* __restrict__ W_hh,
                            unsigned short* __restrict__ hbf0,
                            unsigned short* __restrict__ hT0,
                            unsigned short* __restrict__ Wte_bf,
                            unsigned short* __restrict__ Wih_bf,
                            unsigned short* __restrict__ Whh_bf,
                            float* __restrict__ h_st, float* __restrict__ c_st) {
  const int stride = gridDim.x * blockDim.x;
  for (int i = blockIdx.x * blockDim.x + threadIdx.x; i < EMB * G2 * RNN; i += stride) {
    Wte_bf[i] = f2bf(W_te[i]);  // [64][2048] bf16, no transpose
    if (i < NN * RNN) {
      Wih_bf[i] = f2bf(W_ih[i]);  // [512][128]
      Whh_bf[i] = f2bf(W_hh[i]);  // [512][128]
      float hv = hidden[i];
      h_st[i] = hv;
      c_st[i] = cell[i];
      hbf0[i] = f2bf(hv);                  // node-major [N][RNN]
      int n = i >> 7, r = i & 127;
      hT0[(size_t)r * NN + n] = f2bf(hv);  // transposed [RNN][N]
    }
  }
}

// ---- Social einsum: per node, social = G^T(16x512) x H(512x128) ----
// 256 blocks x 2 nodes: B-tile (hidden state) loaded once, feeds both nodes'
// WMMAs (halves L2 refetch). K split across 8 waves (64 each); partials
// reduced through LDS; bf16 result to social_ws[node][2048].
__global__ __launch_bounds__(256) void social_kernel(
    int t, const float* __restrict__ grids,
    const unsigned short* __restrict__ hT_in,
    unsigned short* __restrict__ social_ws) {
  extern __shared__ unsigned char pool[];
  float* partial = (float*)pool;  // [8][2048] f32 = 64 KB

  const int tid  = threadIdx.x;
  const int wv   = tid >> 5;
  const int lane = tid & 31;
  const int half = lane >> 4;
  const int row  = lane & 15;
  const int nd0  = blockIdx.x * 2;

  const float* G0 = grids + ((size_t)t * NN + nd0) * (size_t)NN * G2;
  const float* G1 = G0 + (size_t)NN * G2;
  const int kbase = wv * 64;

  v8f acc0[8], acc1[8];
#pragma unroll
  for (int ct = 0; ct < 8; ++ct) { acc0[ct] = zero8(); acc1[ct] = zero8(); }

#pragma unroll
  for (int ki = 0; ki < 2; ++ki) {
    const int k0 = kbase + ki * 32;
    if (ki == 0) {
      __builtin_prefetch(&G0[(size_t)(k0 + 32) * G2 + row], 0, 1);
      __builtin_prefetch(&G1[(size_t)(k0 + 32) * G2 + row], 0, 1);
    }
    V16 a0 = gatherA_grid(G0, k0, half, row);
    V16 a1 = gatherA_grid(G1, k0, half, row);
#pragma unroll
    for (int ct = 0; ct < 8; ++ct) {
      V16 b = loadB_row(hT_in, (size_t)(ct * 16 + row) * NN, k0, half);
      acc0[ct] = wmma_bf16(a0, b, acc0[ct]);
      acc1[ct] = wmma_bf16(a1, b, acc1[ct]);
    }
  }

  // reduce node0 then node1 through the same LDS partial buffer
#pragma unroll
  for (int s = 0; s < 2; ++s) {
#pragma unroll
    for (int ct = 0; ct < 8; ++ct)
#pragma unroll
      for (int v = 0; v < 8; ++v)
        partial[wv * 2048 + (v + half * 8) * RNN + ct * 16 + row] =
            (s == 0) ? acc0[ct][v] : acc1[ct][v];
    __syncthreads();
    const int e = tid * 8;
    float sum[8] = {0.f, 0.f, 0.f, 0.f, 0.f, 0.f, 0.f, 0.f};
#pragma unroll
    for (int p = 0; p < 8; ++p)
#pragma unroll
      for (int q = 0; q < 8; ++q) sum[q] += partial[p * 2048 + e + q];
    U8pack o;
#pragma unroll
    for (int q = 0; q < 4; ++q) o.w[q] = pack2bf(sum[2 * q], sum[2 * q + 1]);
    *(US8*)&social_ws[(size_t)(nd0 + s) * 2048 + e] = o.us;
    __syncthreads();
  }
}

// ---- Phases B..E: embeddings, gates, LSTM pointwise, output ----
__global__ __launch_bounds__(256) void step2_kernel(
    int t,
    const float* __restrict__ nodes,
    const float* __restrict__ W_ie, const float* __restrict__ b_ie,
    const float* __restrict__ b_te,
    const float* __restrict__ b_ih, const float* __restrict__ b_hh,
    const float* __restrict__ W_out, const float* __restrict__ b_out,
    const unsigned short* __restrict__ social_ws,
    const unsigned short* __restrict__ hbf_in, unsigned short* __restrict__ hbf_out,
    unsigned short* __restrict__ hT_out,
    const unsigned short* __restrict__ Wte_bf, const unsigned short* __restrict__ Wih_bf,
    const unsigned short* __restrict__ Whh_bf,
    float* __restrict__ outp, float* __restrict__ h_st, float* __restrict__ c_st) {
  __shared__ unsigned short xel[16 * 128];  // 4 KB
  __shared__ float gatesp[16 * 512];        // 32 KB
  __shared__ float hnewp[16 * 128];         // 8 KB

  const int tid  = threadIdx.x;
  const int wv   = tid >> 5;
  const int lane = tid & 31;
  const int half = lane >> 4;
  const int row  = lane & 15;
  const int node_base = blockIdx.x * 16;

  // Phase B: input embedding (VALU) + tensor embedding (WMMA, K=2048)
  for (int e = tid; e < 16 * EMB; e += 256) {
    int ln = e >> 6, j = e & 63;
    const float* x = nodes + ((size_t)t * NN + node_base + ln) * 2;
    float v = W_ie[j * 2 + 0] * x[0] + W_ie[j * 2 + 1] * x[1] + b_ie[j];
    xel[ln * 128 + j] = f2bf(v > 0.f ? v : 0.f);
  }
  if (wv < 4) {
    const int n0 = wv * 16;
    v8f acc = zero8();
    for (int k0 = 0; k0 < G2 * RNN; k0 += 32) {
      V16 a = loadA_row(&social_ws[(size_t)(node_base + row) * 2048], k0, half);
      V16 b = loadB_row(Wte_bf, (size_t)(n0 + row) * (G2 * RNN), k0, half);
      acc = wmma_bf16(a, b, acc);
    }
#pragma unroll
    for (int v = 0; v < 8; ++v) {
      int ln = v + half * 8;
      float val = acc[v] + b_te[n0 + row];
      xel[ln * 128 + EMB + n0 + row] = f2bf(val > 0.f ? val : 0.f);
    }
  }
  __syncthreads();

  // Phase C: gates = xe @ W_ih^T + h @ W_hh^T  (16 x 512, K=128 each)
#pragma unroll
  for (int cc = 0; cc < 4; ++cc) {
    const int ct = wv + cc * 8;
    const int n0 = ct * 16;
    v8f acc = zero8();
#pragma unroll
    for (int k0 = 0; k0 < 2 * EMB; k0 += 32) {
      V16 a = loadA_row(&xel[row * 128], k0, half);
      V16 b = loadB_row(Wih_bf, (size_t)(n0 + row) * (2 * EMB), k0, half);
      acc = wmma_bf16(a, b, acc);
    }
#pragma unroll
    for (int k0 = 0; k0 < RNN; k0 += 32) {
      V16 a = loadA_row(&hbf_in[(size_t)(node_base + row) * RNN], k0, half);
      V16 b = loadB_row(Whh_bf, (size_t)(n0 + row) * RNN, k0, half);
      acc = wmma_bf16(a, b, acc);
    }
#pragma unroll
    for (int v = 0; v < 8; ++v)
      gatesp[(v + half * 8) * (4 * RNN) + n0 + row] = acc[v];
  }
  __syncthreads();

  // Phase D: LSTM pointwise; h/c update (both h layouts for next step)
  for (int idx = tid; idx < 16 * RNN; idx += 256) {
    int ln = idx >> 7, r = idx & 127;
    int gn = node_base + ln;
    float i_ = gatesp[ln * 512 + r]       + b_ih[r]       + b_hh[r];
    float f_ = gatesp[ln * 512 + 128 + r] + b_ih[128 + r] + b_hh[128 + r];
    float g_ = gatesp[ln * 512 + 256 + r] + b_ih[256 + r] + b_hh[256 + r];
    float o_ = gatesp[ln * 512 + 384 + r] + b_ih[384 + r] + b_hh[384 + r];
    float si = 1.f / (1.f + __expf(-i_));
    float sf = 1.f / (1.f + __expf(-f_));
    float so = 1.f / (1.f + __expf(-o_));
    float cn = sf * c_st[(size_t)gn * RNN + r] + si * tanhf(g_);
    float hn = so * tanhf(cn);
    c_st[(size_t)gn * RNN + r] = cn;
    h_st[(size_t)gn * RNN + r] = hn;
    unsigned short hb = f2bf(hn);
    hbf_out[(size_t)gn * RNN + r] = hb;
    hT_out[(size_t)r * NN + gn]   = hb;
    hnewp[ln * RNN + r] = hn;
  }
  __syncthreads();

  // Phase E: out = h_new @ W_out^T + b_out
  if (tid < 16 * OUTD) {
    int ln = tid / OUTD, o = tid % OUTD;
    float s = b_out[o];
#pragma unroll 8
    for (int k = 0; k < RNN; ++k) s += hnewp[ln * RNN + k] * W_out[o * RNN + k];
    outp[((size_t)t * NN + node_base + ln) * OUTD + o] = s;
  }
}

extern "C" void kernel_launch(void* const* d_in, const int* in_sizes, int n_in,
                              void* d_out, int out_size, void* d_ws, size_t ws_size,
                              hipStream_t stream) {
  (void)in_sizes; (void)n_in; (void)out_size; (void)ws_size;
  const float* nodes  = (const float*)d_in[0];
  const float* grids  = (const float*)d_in[1];
  /* d_in[2] nodesPresent: all nodes present -> unused */
  const float* hidden = (const float*)d_in[3];
  const float* cell   = (const float*)d_in[4];
  const float* W_ie   = (const float*)d_in[5];
  const float* b_ie   = (const float*)d_in[6];
  const float* W_te   = (const float*)d_in[7];
  const float* b_te   = (const float*)d_in[8];
  const float* W_ih   = (const float*)d_in[9];
  const float* W_hh   = (const float*)d_in[10];
  const float* b_ih   = (const float*)d_in[11];
  const float* b_hh   = (const float*)d_in[12];
  const float* W_out  = (const float*)d_in[13];
  const float* b_out  = (const float*)d_in[14];

  float* outp = (float*)d_out;                   // [SEQ][N][OUT]
  float* h_st = outp + (size_t)SEQ * NN * OUTD;  // h_fin [N][RNN]
  float* c_st = h_st + (size_t)NN * RNN;         // c_fin [N][RNN]

  unsigned char* ws = (unsigned char*)d_ws;
  unsigned short* hbf0   = (unsigned short*)(ws);             // 128 KB node-major h
  unsigned short* hbf1   = (unsigned short*)(ws + 131072);    // 128 KB
  unsigned short* hT0    = (unsigned short*)(ws + 262144);    // 128 KB transposed h
  unsigned short* hT1    = (unsigned short*)(ws + 393216);    // 128 KB
  unsigned short* Wte_bf = (unsigned short*)(ws + 524288);    // 256 KB [64][2048]
  unsigned short* Wih_bf = (unsigned short*)(ws + 786432);    // 128 KB [512][128]
  unsigned short* Whh_bf = (unsigned short*)(ws + 917504);    // 128 KB [512][128]
  unsigned short* social = (unsigned short*)(ws + 1048576);   // 2 MB [512][2048] bf16

  init_kernel<<<dim3(256), dim3(256), 0, stream>>>(
      hidden, cell, W_te, W_ih, W_hh, hbf0, hT0, Wte_bf, Wih_bf, Whh_bf, h_st, c_st);

  for (int t = 0; t < SEQ; ++t) {
    unsigned short* hin   = (t & 1) ? hbf1 : hbf0;
    unsigned short* hout  = (t & 1) ? hbf0 : hbf1;
    unsigned short* hTin  = (t & 1) ? hT1 : hT0;
    unsigned short* hTout = (t & 1) ? hT0 : hT1;
    social_kernel<<<dim3(NN / 2), dim3(256), 65536, stream>>>(t, grids, hTin, social);
    step2_kernel<<<dim3(NN / 16), dim3(256), 0, stream>>>(
        t, nodes, W_ie, b_ie, b_te, b_ih, b_hh, W_out, b_out,
        social, hin, hout, hTout, Wte_bf, Wih_bf, Whh_bf, outp, h_st, c_st);
  }
}
